// PolyPrecond_30039001268521
// MI455X (gfx1250) — compile-verified, matrix operand order checked by reading, not run on hardware
//
#include <hip/hip_runtime.h>
#include <hip/hip_bf16.h>

typedef _Float16 h16;
typedef __attribute__((ext_vector_type(16))) _Float16 v16h;
typedef __attribute__((ext_vector_type(8)))  _Float16 v8h;
typedef __attribute__((ext_vector_type(8)))  float    v8f;
typedef __attribute__((ext_vector_type(4)))  int      v4i;

// ---------------------------------------------------------------------------
// WMMA helpers (CDNA5 gfx1250, wave32)
// A 16x32 f16 layout: lane L holds row m=L&15; K elements:
//   k = (L>>4)*8 + (e&7) + (e>>3)*16  for element e of the v16h.
// B fragments are pre-packed so lane L reads 16 contiguous halfs.
// C/D 16x16 f32 layout: element r of v8f is (m = r + 8*(L>>4), n = L&15).
// ---------------------------------------------------------------------------

__device__ __forceinline__ v16h load_a_frag(const h16* p, int lane) {
    int kb = (lane >> 4) * 8;
    v8h lo = *(const v8h*)(p + kb);
    v8h hi = *(const v8h*)(p + kb + 16);
    v16h a;
#pragma unroll
    for (int i = 0; i < 8; ++i) { a[i] = lo[i]; a[i + 8] = hi[i]; }
    return a;
}

__device__ __forceinline__ v16h load_b_frag(const h16* frag, int lane) {
    const h16* p = frag + lane * 16;
    v8h lo = *(const v8h*)(p);
    v8h hi = *(const v8h*)(p + 8);
    v16h b;
#pragma unroll
    for (int i = 0; i < 8; ++i) { b[i] = lo[i]; b[i + 8] = hi[i]; }
    return b;
}

__device__ __forceinline__ v8f wmma_f32_f16(v16h a, v16h b, v8f c) {
    return __builtin_amdgcn_wmma_f32_16x16x32_f16(false, a, false, b,
                                                  (short)0, c, false, false);
}

// ---------------------------------------------------------------------------
// Async global->LDS copy (CDNA5 GLOBAL_LOAD_ASYNC_TO_LDS_B128, ASYNCcnt).
// Builtin signature (from hipcc diagnostic): (v4i AS1*, v4i AS3*, imm, imm).
// Falls back to load + ds_store if the builtin is unavailable.
// ---------------------------------------------------------------------------
typedef __attribute__((address_space(1))) v4i* gas_v4i;
typedef __attribute__((address_space(3))) v4i* las_v4i;

__device__ __forceinline__ void async_copy_16B(const h16* g, h16* l) {
#if __has_builtin(__builtin_amdgcn_global_load_async_to_lds_b128)
    __builtin_amdgcn_global_load_async_to_lds_b128((gas_v4i)g, (las_v4i)l, 0, 0);
#else
    *(v8h*)l = *(const v8h*)g;
#endif
}

__device__ __forceinline__ void async_wait0() {
#if __has_builtin(__builtin_amdgcn_s_wait_asynccnt)
    __builtin_amdgcn_s_wait_asynccnt(0);
#elif __has_builtin(__builtin_amdgcn_global_load_async_to_lds_b128)
    asm volatile("s_wait_asynccnt 0" ::: "memory");
#endif
}

// cooperative block staging of packed weights into LDS (16B per thread-step)
__device__ __forceinline__ void stage_weights(const h16* __restrict__ g,
                                              h16* l, int halfs, int tid) {
    for (int i = tid * 8; i < halfs; i += 256 * 8) async_copy_16B(g + i, l + i);
}

// ---------------------------------------------------------------------------
// Weight pre-pack: f32 row-major W[K][N] -> f16 WMMA-B fragment order.
// frag index = kt*(N/16)+nt ; within fragment: [lane][e] with
//   k = kt*32 + (lane>>4)*8 + (e&7) + (e>>3)*16 ; n = nt*16 + (lane&15)
// ---------------------------------------------------------------------------
__global__ void pack_b_kernel(const float* __restrict__ W, h16* __restrict__ out,
                              int K, int N) {
    int total = K * N;
    for (int o = blockIdx.x * blockDim.x + threadIdx.x; o < total;
         o += gridDim.x * blockDim.x) {
        int e    = o & 15;
        int lane = (o >> 4) & 31;
        int fr   = o >> 9;
        int nTiles = N >> 4;
        int nt = fr % nTiles;
        int kt = fr / nTiles;
        int k = kt * 32 + ((lane >> 4) * 8) + (e & 7) + ((e >> 3) * 16);
        int n = nt * 16 + (lane & 15);
        out[o] = (h16)W[k * N + n];
    }
}

// ---------------------------------------------------------------------------
// Small-input 2-layer MLP encoder (f32 math), writes f16 latents.
// One block = 64 threads, processes rowsPerBlock rows; weights staged in LDS.
// ---------------------------------------------------------------------------
template <int DIN>
__global__ __launch_bounds__(64) void encode_kernel(
    const float* __restrict__ X,
    const float* __restrict__ W1, const float* __restrict__ B1,
    const float* __restrict__ W2, const float* __restrict__ B2,
    h16* __restrict__ out0, h16* __restrict__ out1, float* __restrict__ outF,
    int R, int rowsPerBlock) {
    __shared__ float sW1[DIN * 64];
    __shared__ float sW2[64 * 64];
    __shared__ float xin[DIN];
    __shared__ float h[64];
    int j = threadIdx.x;
    for (int i = j; i < DIN * 64; i += 64) sW1[i] = W1[i];
    for (int i = j; i < 64 * 64; i += 64) sW2[i] = W2[i];
    __syncthreads();
    float b1 = B1[j], b2 = B2[j];
    long r0 = (long)blockIdx.x * rowsPerBlock;
    for (int rr = 0; rr < rowsPerBlock; ++rr) {
        long r = r0 + rr;
        if (r >= R) break;  // uniform across block
        if (j < DIN) xin[j] = X[r * DIN + j];
        __syncthreads();
        float acc = b1;
#pragma unroll
        for (int i = 0; i < DIN; ++i) acc += xin[i] * sW1[i * 64 + j];
        h[j] = fmaxf(acc, 0.0f);
        __syncthreads();
        float o = b2;
        for (int k = 0; k < 64; ++k) o += h[k] * sW2[k * 64 + j];
        out0[r * 64 + j] = (h16)o;
        if (out1) out1[r * 64 + j] = (h16)o;
        if (outF) outF[r * 64 + j] = o;
        __syncthreads();
    }
}

// degree = segment_sum(ones, dst)
__global__ void deg_kernel(const int* __restrict__ dst, float* __restrict__ deg,
                           int E) {
    int i = blockIdx.x * blockDim.x + threadIdx.x;
    if (i < E) atomicAdd(&deg[dst[i]], 1.0f);
}

// aggH = (aggF / max(deg,1)) as f16
__global__ void finalize_agg_kernel(const float* __restrict__ aggF,
                                    const float* __restrict__ deg,
                                    h16* __restrict__ aggH, long total) {
    long i = (long)blockIdx.x * blockDim.x + threadIdx.x;
    if (i < total) {
        float dg = fmaxf(deg[i >> 6], 1.0f);
        aggH[i] = (h16)(aggF[i] / dg);
    }
}

// ---------------------------------------------------------------------------
// Fused edge update: e_in = skip([edge, edge0]); edge' = MLP([e_in, n_src,
// n_dst, g]); accumulate segment-sum into aggF and edge feature sums.
// 1 wave per 16-edge tile, 8 waves/block; weights staged in LDS (async).
// ---------------------------------------------------------------------------
__global__ __launch_bounds__(256) void edge_layer_kernel(
    const h16* __restrict__ edgeIn, const h16* __restrict__ edge0,
    const h16* __restrict__ nodeIn, const h16* __restrict__ gH,
    const int* __restrict__ srcIdx, const int* __restrict__ dstIdx,
    const h16* __restrict__ sW, const float* __restrict__ sB,
    const h16* __restrict__ W1, const float* __restrict__ B1,
    const h16* __restrict__ W2, const float* __restrict__ B2,
    h16* __restrict__ edgeOut, float* __restrict__ aggF,
    float* __restrict__ edgeSum, int numE) {
    __shared__ __align__(16) h16 Ein[8][16 * 64];
    __shared__ __align__(16) h16 Hs[8][16 * 64];
    __shared__ __align__(16) h16 sWs[8192];    // 128x64 skip
    __shared__ __align__(16) h16 W1s[16384];   // 256x64
    __shared__ __align__(16) h16 W2s[4096];    // 64x64
    __shared__ int dstT[8][16];
    __shared__ float esum[64];

    int tid = threadIdx.x;
    // kick off async weight staging (overlaps address setup below)
    stage_weights(sW, sWs, 8192, tid);
    stage_weights(W1, W1s, 16384, tid);
    stage_weights(W2, W2s, 4096, tid);

    if (tid < 64) esum[tid] = 0.0f;
    int w = tid >> 5, lane = tid & 31;
    int m = lane & 15, hi = lane >> 4, n = m;
    long tileBase = (long)(blockIdx.x * 8 + w) * 16;
    long rowIdx = tileBase + m;
    long rowC = rowIdx < numE ? rowIdx : (long)(numE - 1);
    int s = srcIdx[rowC];
    int d = dstIdx[rowC];
    if (hi == 0) dstT[w][m] = d;
    const h16* pe = edgeIn + rowC * 64;
    const h16* p0 = edge0 + rowC * 64;
    const h16* ps = nodeIn + (long)s * 64;
    const h16* pd = nodeIn + (long)d * 64;
    async_wait0();
    __syncthreads();

    // ---- Phase 1: skip linear, K = 128 : [edge | edge0] @ sW + sB ----------
#pragma unroll
    for (int nt = 0; nt < 4; ++nt) {
        v8f c;
        float bv = sB[nt * 16 + n];
#pragma unroll
        for (int r = 0; r < 8; ++r) c[r] = bv;
#pragma unroll
        for (int kt = 0; kt < 4; ++kt) {
            const h16* ap = (kt < 2) ? (pe + kt * 32) : (p0 + (kt - 2) * 32);
            v16h a = load_a_frag(ap, lane);
            v16h b = load_b_frag(sWs + (kt * 4 + nt) * 512, lane);
            c = wmma_f32_f16(a, b, c);
        }
#pragma unroll
        for (int r = 0; r < 8; ++r)
            Ein[w][(r + hi * 8) * 64 + nt * 16 + n] = (h16)c[r];
    }
    __syncthreads();

    // ---- Phase 2: edge MLP layer 1, K = 256 : [e_in|n_src|n_dst|g] ---------
    const h16* einRow = &Ein[w][m * 64];
#pragma unroll
    for (int nt = 0; nt < 4; ++nt) {
        v8f c;
        float bv = B1[nt * 16 + n];
#pragma unroll
        for (int r = 0; r < 8; ++r) c[r] = bv;
#pragma unroll
        for (int kt = 0; kt < 8; ++kt) {
            int kg = kt * 32;
            const h16* ap;
            if (kg < 64)        ap = einRow + kg;
            else if (kg < 128)  ap = ps + (kg - 64);
            else if (kg < 192)  ap = pd + (kg - 128);
            else                ap = gH + (kg - 192);
            v16h a = load_a_frag(ap, lane);
            v16h b = load_b_frag(W1s + (kt * 4 + nt) * 512, lane);
            c = wmma_f32_f16(a, b, c);
        }
#pragma unroll
        for (int r = 0; r < 8; ++r)
            Hs[w][(r + hi * 8) * 64 + nt * 16 + n] = (h16)fmaxf(c[r], 0.0f);
    }
    __syncthreads();

    // ---- Phase 3: edge MLP layer 2, K = 64; store + segment-sum ------------
    const h16* hRow = &Hs[w][m * 64];
#pragma unroll
    for (int nt = 0; nt < 4; ++nt) {
        v8f c;
        float bv = B2[nt * 16 + n];
#pragma unroll
        for (int r = 0; r < 8; ++r) c[r] = bv;
#pragma unroll
        for (int kt = 0; kt < 2; ++kt) {
            v16h a = load_a_frag(hRow + kt * 32, lane);
            v16h b = load_b_frag(W2s + (kt * 4 + nt) * 512, lane);
            c = wmma_f32_f16(a, b, c);
        }
        int nG = nt * 16 + n;
#pragma unroll
        for (int r = 0; r < 8; ++r) {
            long mG = tileBase + r + hi * 8;
            if (mG < numE) {
                float v = c[r];
                edgeOut[mG * 64 + nG] = (h16)v;
                int dd = dstT[w][r + hi * 8];
                atomicAdd(&aggF[(long)dd * 64 + nG], v);
                atomicAdd(&esum[nG], v);  // LDS ds_add_f32
            }
        }
    }
    __syncthreads();
    if (tid < 64) atomicAdd(&edgeSum[tid], esum[tid]);
}

// ---------------------------------------------------------------------------
// Fused node update: node' = MLP([node, agg, g]), K = 192; node feature sums.
// ---------------------------------------------------------------------------
__global__ __launch_bounds__(256) void node_layer_kernel(
    const h16* __restrict__ nodeIn, const h16* __restrict__ aggH,
    const h16* __restrict__ gH,
    const h16* __restrict__ W1, const float* __restrict__ B1,
    const h16* __restrict__ W2, const float* __restrict__ B2,
    h16* __restrict__ nodeOut, float* __restrict__ nodeSum, int numN) {
    __shared__ __align__(16) h16 Hs[8][16 * 64];
    __shared__ __align__(16) h16 W1s[12288];   // 192x64
    __shared__ __align__(16) h16 W2s[4096];    // 64x64
    __shared__ float nsum[64];
    int tid = threadIdx.x;
    stage_weights(W1, W1s, 12288, tid);
    stage_weights(W2, W2s, 4096, tid);
    if (tid < 64) nsum[tid] = 0.0f;
    int w = tid >> 5, lane = tid & 31;
    int m = lane & 15, hi = lane >> 4, n = m;
    long tileBase = (long)(blockIdx.x * 8 + w) * 16;
    long rowIdx = tileBase + m;
    long rowC = rowIdx < numN ? rowIdx : (long)(numN - 1);
    const h16* pn = nodeIn + rowC * 64;
    const h16* pa = aggH + rowC * 64;
    async_wait0();
    __syncthreads();

#pragma unroll
    for (int nt = 0; nt < 4; ++nt) {
        v8f c;
        float bv = B1[nt * 16 + n];
#pragma unroll
        for (int r = 0; r < 8; ++r) c[r] = bv;
#pragma unroll
        for (int kt = 0; kt < 6; ++kt) {
            const h16* ap;
            if (kt < 2)      ap = pn + kt * 32;
            else if (kt < 4) ap = pa + (kt - 2) * 32;
            else             ap = gH + (kt - 4) * 32;
            v16h a = load_a_frag(ap, lane);
            v16h b = load_b_frag(W1s + (kt * 4 + nt) * 512, lane);
            c = wmma_f32_f16(a, b, c);
        }
#pragma unroll
        for (int r = 0; r < 8; ++r)
            Hs[w][(r + hi * 8) * 64 + nt * 16 + n] = (h16)fmaxf(c[r], 0.0f);
    }
    __syncthreads();

    const h16* hRow = &Hs[w][m * 64];
#pragma unroll
    for (int nt = 0; nt < 4; ++nt) {
        v8f c;
        float bv = B2[nt * 16 + n];
#pragma unroll
        for (int r = 0; r < 8; ++r) c[r] = bv;
#pragma unroll
        for (int kt = 0; kt < 2; ++kt) {
            v16h a = load_a_frag(hRow + kt * 32, lane);
            v16h b = load_b_frag(W2s + (kt * 4 + nt) * 512, lane);
            c = wmma_f32_f16(a, b, c);
        }
        int nG = nt * 16 + n;
#pragma unroll
        for (int r = 0; r < 8; ++r) {
            long mG = tileBase + r + hi * 8;
            if (mG < numN) {
                float v = c[r];
                nodeOut[mG * 64 + nG] = (h16)v;
                atomicAdd(&nsum[nG], v);
            }
        }
    }
    __syncthreads();
    if (tid < 64) atomicAdd(&nodeSum[tid], nsum[tid]);
}

// g' = MLP([g, mean(node), mean(edge)]) — tiny, f32 scalar, one block.
__global__ __launch_bounds__(64) void global_update_kernel(
    float* __restrict__ gF, h16* __restrict__ gH,
    const float* __restrict__ nodeSum, const float* __restrict__ edgeSum,
    const float* __restrict__ W1, const float* __restrict__ B1,
    const float* __restrict__ W2, const float* __restrict__ B2,
    float invN, float invE) {
    __shared__ float in[192];
    __shared__ float h[64];
    int j = threadIdx.x;
    in[j] = gF[j];
    in[64 + j] = nodeSum[j] * invN;
    in[128 + j] = edgeSum[j] * invE;
    __syncthreads();
    float acc = B1[j];
    for (int i = 0; i < 192; ++i) acc += in[i] * W1[i * 64 + j];
    h[j] = fmaxf(acc, 0.0f);
    __syncthreads();
    float o = B2[j];
    for (int k = 0; k < 64; ++k) o += h[k] * W2[k * 64 + j];
    gF[j] = o;
    gH[j] = (h16)o;
}

// decoder: [mean(node), g] -> relu lin -> relu lin -> 8 raw = 4 complex coeffs
__global__ __launch_bounds__(64) void decode_kernel(
    const float* __restrict__ nodeSum, const float* __restrict__ gF,
    const float* __restrict__ W1, const float* __restrict__ B1,
    const float* __restrict__ W2, const float* __restrict__ B2,
    const float* __restrict__ W3, const float* __restrict__ B3,
    float* __restrict__ out, float invN) {
    __shared__ float in[128];
    __shared__ float h1[64];
    __shared__ float h2[64];
    int j = threadIdx.x;
    in[j] = nodeSum[j] * invN;
    in[64 + j] = gF[j];
    __syncthreads();
    float a = B1[j];
    for (int i = 0; i < 128; ++i) a += in[i] * W1[i * 64 + j];
    h1[j] = fmaxf(a, 0.0f);
    __syncthreads();
    float b = B2[j];
    for (int k = 0; k < 64; ++k) b += h1[k] * W2[k * 64 + j];
    h2[j] = fmaxf(b, 0.0f);
    __syncthreads();
    if (j < 8) {
        float o = B3[j];
        for (int k = 0; k < 64; ++k) o += h2[k] * W3[k * 8 + j];
        out[j] = o;  // interleaved re/im == complex64 flat
    }
}

// ---------------------------------------------------------------------------
// Host-side orchestration
// ---------------------------------------------------------------------------
extern "C" void kernel_launch(void* const* d_in, const int* in_sizes, int n_in,
                              void* d_out, int out_size, void* d_ws, size_t ws_size,
                              hipStream_t stream) {
    const int N = in_sizes[0] / 9;
    const int E = in_sizes[1] / 2;

    const float* x     = (const float*)d_in[0];
    const int*   eidx  = (const int*)d_in[1];
    const float* eattr = (const float*)d_in[2];
    const float* gfeat = (const float*)d_in[3];
    const int* src = eidx;
    const int* dst = eidx + E;

    // params (jax pytree: dict keys sorted)
    const float* dec1W = (const float*)d_in[4];
    const float* dec1B = (const float*)d_in[5];
    const float* dec2W = (const float*)d_in[6];
    const float* dec2B = (const float*)d_in[7];
    const float* dec3W = (const float*)d_in[8];
    const float* dec3B = (const float*)d_in[9];
    const float* eeW1 = (const float*)d_in[10];
    const float* eeB1 = (const float*)d_in[11];
    const float* eeW2 = (const float*)d_in[12];
    const float* eeB2 = (const float*)d_in[13];
    const float* geW1 = (const float*)d_in[14];
    const float* geB1 = (const float*)d_in[15];
    const float* geW2 = (const float*)d_in[16];
    const float* geB2 = (const float*)d_in[17];
    // layers: base = 18 + 14*l : edge_mlp(4) global_mlp(4) node_mlp(4) skip(2)
    const float* neW1 = (const float*)d_in[18 + 14 * 6 + 0];
    const float* neB1 = (const float*)d_in[18 + 14 * 6 + 1];
    const float* neW2 = (const float*)d_in[18 + 14 * 6 + 2];
    const float* neB2 = (const float*)d_in[18 + 14 * 6 + 3];

    // workspace bump allocator
    char* p = (char*)d_ws;
    auto alloc = [&](size_t bytes) -> void* {
        char* r = p;
        p += (bytes + 255) & ~(size_t)255;
        return (void*)r;
    };
    h16* nodeH[2] = {(h16*)alloc((size_t)N * 64 * 2),
                     (h16*)alloc((size_t)N * 64 * 2)};
    h16* edgeH[2] = {(h16*)alloc((size_t)E * 64 * 2),
                     (h16*)alloc((size_t)E * 64 * 2)};
    h16* edge0H = (h16*)alloc((size_t)E * 64 * 2);
    float* aggF = (float*)alloc((size_t)N * 64 * 4);
    h16* aggH = (h16*)alloc((size_t)N * 64 * 2);
    float* deg = (float*)alloc((size_t)N * 4);
    float* gF = (float*)alloc(64 * 4);
    h16* gH = (h16*)alloc(64 * 2);
    float* edgeSum = (float*)alloc(64 * 4);
    float* nodeSum = (float*)alloc(64 * 4);
    // packed weights: per layer [sW 8192 | eW1 16384 | eW2 4096 | nW1 12288 | nW2 4096]
    const size_t layerHalfs = 8192 + 16384 + 4096 + 12288 + 4096;
    h16* wPack = (h16*)alloc(layerHalfs * 6 * 2);

    // ---- weight packing -----------------------------------------------------
    for (int l = 0; l < 6; ++l) {
        const int base = 18 + 14 * l;
        const float* emW1 = (const float*)d_in[base + 0];
        const float* emW2 = (const float*)d_in[base + 2];
        const float* nmW1 = (const float*)d_in[base + 8];
        const float* nmW2 = (const float*)d_in[base + 10];
        const float* skW  = (const float*)d_in[base + 12];
        h16* lp = wPack + (size_t)l * layerHalfs;
        pack_b_kernel<<<32, 256, 0, stream>>>(skW, lp, 128, 64);
        pack_b_kernel<<<64, 256, 0, stream>>>(emW1, lp + 8192, 256, 64);
        pack_b_kernel<<<16, 256, 0, stream>>>(emW2, lp + 8192 + 16384, 64, 64);
        pack_b_kernel<<<48, 256, 0, stream>>>(nmW1, lp + 8192 + 16384 + 4096, 192, 64);
        pack_b_kernel<<<16, 256, 0, stream>>>(nmW2, lp + 8192 + 16384 + 4096 + 12288, 64, 64);
    }

    // ---- degrees ------------------------------------------------------------
    (void)hipMemsetAsync(deg, 0, (size_t)N * 4, stream);
    deg_kernel<<<(E + 255) / 256, 256, 0, stream>>>(dst, deg, E);

    // ---- encoders -----------------------------------------------------------
    encode_kernel<9><<<(N + 31) / 32, 64, 0, stream>>>(
        x, neW1, neB1, neW2, neB2, nodeH[0], nullptr, nullptr, N, 32);
    encode_kernel<8><<<(E + 31) / 32, 64, 0, stream>>>(
        eattr, eeW1, eeB1, eeW2, eeB2, edgeH[0], edge0H, nullptr, E, 32);
    encode_kernel<4><<<1, 64, 0, stream>>>(
        gfeat, geW1, geB1, geW2, geB2, gH, nullptr, gF, 1, 1);

    // ---- message passing ----------------------------------------------------
    const int edgeBlocks = (E + 127) / 128;
    const int nodeBlocks = (N + 127) / 128;
    const float invN = 1.0f / (float)N;
    const float invE = 1.0f / (float)E;
    for (int l = 0; l < 6; ++l) {
        const int base = 18 + 14 * l;
        const float* emB1 = (const float*)d_in[base + 1];
        const float* emB2 = (const float*)d_in[base + 3];
        const float* gmW1 = (const float*)d_in[base + 4];
        const float* gmB1 = (const float*)d_in[base + 5];
        const float* gmW2 = (const float*)d_in[base + 6];
        const float* gmB2 = (const float*)d_in[base + 7];
        const float* nmB1 = (const float*)d_in[base + 9];
        const float* nmB2 = (const float*)d_in[base + 11];
        const float* skB  = (const float*)d_in[base + 13];
        h16* lp = wPack + (size_t)l * layerHalfs;
        h16* sW  = lp;
        h16* eW1 = lp + 8192;
        h16* eW2 = lp + 8192 + 16384;
        h16* nW1 = lp + 8192 + 16384 + 4096;
        h16* nW2 = lp + 8192 + 16384 + 4096 + 12288;
        h16* eIn = edgeH[l & 1];
        h16* eOut = edgeH[(l + 1) & 1];
        h16* nIn = nodeH[l & 1];
        h16* nOut = nodeH[(l + 1) & 1];

        (void)hipMemsetAsync(aggF, 0, (size_t)N * 64 * 4, stream);
        (void)hipMemsetAsync(edgeSum, 0, 64 * 4, stream);
        (void)hipMemsetAsync(nodeSum, 0, 64 * 4, stream);

        edge_layer_kernel<<<edgeBlocks, 256, 0, stream>>>(
            eIn, edge0H, nIn, gH, src, dst, sW, skB, eW1, emB1, eW2, emB2,
            eOut, aggF, edgeSum, E);

        finalize_agg_kernel<<<(int)(((size_t)N * 64 + 255) / 256), 256, 0, stream>>>(
            aggF, deg, aggH, (long)N * 64);

        node_layer_kernel<<<nodeBlocks, 256, 0, stream>>>(
            nIn, aggH, gH, nW1, nmB1, nW2, nmB2, nOut, nodeSum, N);

        global_update_kernel<<<1, 64, 0, stream>>>(
            gF, gH, nodeSum, edgeSum, gmW1, gmB1, gmW2, gmB2, invN, invE);
    }

    // ---- decode -------------------------------------------------------------
    decode_kernel<<<1, 64, 0, stream>>>(nodeSum, gF, dec1W, dec1B, dec2W, dec2B,
                                        dec3W, dec3B, (float*)d_out, invN);
}
